// DifferentiableSVM_32942399160660
// MI455X (gfx1250) — compile-verified, matrix operand order checked by reading, not run on hardware
//
#include <hip/hip_runtime.h>
#include <hip/hip_bf16.h>
#include <stdint.h>

// ---------------------------------------------------------------------------
// Problem constants (fixed by setup_inputs)
// ---------------------------------------------------------------------------
#define NS 4096          // n_support
#define ED 2048          // embed_dim (bias handled separately)
#define NQ 16384         // n_query
#define NC 128           // n_classes
#define MAX_ITER 15
#define LRATE 0.01f
#define CREG 1.0f

#define AS1 __attribute__((address_space(1)))
#define AS3 __attribute__((address_space(3)))

typedef __attribute__((ext_vector_type(16))) __bf16       v16bf;
typedef __attribute__((ext_vector_type(8)))  float        v8f;
typedef __attribute__((ext_vector_type(4)))  unsigned int u32x4;
typedef __attribute__((ext_vector_type(2)))  unsigned int u32x2;
typedef __attribute__((ext_vector_type(4)))  float        f32x4;
typedef int v4i __attribute__((vector_size(16)));   // matches async-builtin param

#if defined(__has_builtin)
#  if __has_builtin(__builtin_amdgcn_global_load_async_to_lds_b128)
#    define HAVE_ASYNC_COPY 1
#  endif
#  if __has_builtin(__builtin_amdgcn_s_wait_asynccnt)
#    define HAVE_ASYNC_WAIT 1
#  endif
#endif
#ifndef HAVE_ASYNC_COPY
#  define HAVE_ASYNC_COPY 0
#endif
#ifndef HAVE_ASYNC_WAIT
#  define HAVE_ASYNC_WAIT 0
#endif

union FragAB {          // one 16x32 (A) or 32x16 (B) bf16 WMMA fragment
    v16bf        bf;
    u32x4        q[2];
    unsigned int u[8];
};

// f32 -> bf16 (round to nearest even)
__device__ __forceinline__ unsigned int bf16_1(float f) {
    unsigned int u = __builtin_bit_cast(unsigned int, f);
    u += 0x7FFFu + ((u >> 16) & 1u);
    return u >> 16;
}
__device__ __forceinline__ unsigned int bf16_2(float lo, float hi) {
    unsigned int ul = __builtin_bit_cast(unsigned int, lo);
    unsigned int uh = __builtin_bit_cast(unsigned int, hi);
    ul += 0x7FFFu + ((ul >> 16) & 1u);
    uh += 0x7FFFu + ((uh >> 16) & 1u);
    return (ul >> 16) | (uh & 0xFFFF0000u);
}
__device__ __forceinline__ float bf16_to_f(unsigned int h16) {
    unsigned int u = h16 << 16;
    return __builtin_bit_cast(float, u);
}

__device__ __forceinline__ v8f wmma_bf16(const v16bf& a, const v16bf& b, v8f c) {
    return __builtin_amdgcn_wmma_f32_16x16x32_bf16(
        false, a, false, b, (short)0, c, false, false);
}

__device__ __forceinline__ void async_copy_wait() {
#if HAVE_ASYNC_COPY
#  if HAVE_ASYNC_WAIT
    __builtin_amdgcn_s_wait_asynccnt(0);
#  else
    asm volatile("s_wait_asynccnt 0x0" ::: "memory");
#  endif
#endif
}

// Stage one 32-K x 128-class bf16 slab of W^T (class-major) into LDS.
// NTHR threads: each copies NC/NTHR class rows (64 B each = 4 x b128).
// Uses gfx1250 GLOBAL_LOAD_ASYNC_TO_LDS_B128 (ASYNCcnt) so the copy overlaps
// WMMA work on the other LDS buffer.
template <int NTHR>
__device__ __forceinline__ void stage_slab(const unsigned short* __restrict__ wb,
                                           int k0,
                                           unsigned short (*dst)[32],
                                           int tid) {
    #pragma unroll
    for (int r = tid; r < NC; r += NTHR) {
        const u32x4* g = (const u32x4*)(wb + (size_t)r * ED + k0);
        u32x4*       l = (u32x4*)(&dst[r][0]);
#if HAVE_ASYNC_COPY
        AS1 v4i* gp = (AS1 v4i*)g;
        AS3 v4i* lp = (AS3 v4i*)l;
        __builtin_amdgcn_global_load_async_to_lds_b128(gp, lp,  0, 0);
        __builtin_amdgcn_global_load_async_to_lds_b128(gp, lp, 16, 0);
        __builtin_amdgcn_global_load_async_to_lds_b128(gp, lp, 32, 0);
        __builtin_amdgcn_global_load_async_to_lds_b128(gp, lp, 48, 0);
#else
        #pragma unroll
        for (int c = 0; c < 4; ++c) l[c] = g[c];
#endif
    }
}

// ---------------------------------------------------------------------------
// Setup kernels
// ---------------------------------------------------------------------------
__global__ void k_zero_w(float* __restrict__ wt, unsigned short* __restrict__ wbb,
                         float* __restrict__ bias) {
    int i = blockIdx.x * blockDim.x + threadIdx.x;
    if (i < NC * ED) { wt[i] = 0.0f; wbb[i] = 0; }
    if (i < NC)      bias[i] = 0.0f;
}

// support f32 [NS][ED] -> Xb bf16 [NS][ED] and XTb bf16 [ED][NS]
// 32x32 tile transpose through LDS so both outputs store coalesced (8B).
__global__ __launch_bounds__(256) void k_cvt_support(
    const float* __restrict__ sup,
    unsigned short* __restrict__ xb,
    unsigned short* __restrict__ xtb) {
    __shared__ unsigned short tile[32][33];
    const int bi = blockIdx.x & (NS / 32 - 1);   // 128 i-tiles
    const int bk = blockIdx.x >> 7;              // 64  k-tiles
    const int i0 = bi * 32, k0 = bk * 32;
    const int tid = threadIdx.x;

    {   // load + convert + row-major store (4 consecutive k per thread)
        int r = tid >> 3;             // i within tile: 0..31
        int c = (tid & 7) * 4;        // k within tile: 0,4,..,28
        f32x4 v = *(const f32x4*)(sup + (size_t)(i0 + r) * ED + k0 + c);
        unsigned int h0 = bf16_1(v.x), h1 = bf16_1(v.y);
        unsigned int h2 = bf16_1(v.z), h3 = bf16_1(v.w);
        tile[r][c + 0] = (unsigned short)h0;  tile[r][c + 1] = (unsigned short)h1;
        tile[r][c + 2] = (unsigned short)h2;  tile[r][c + 3] = (unsigned short)h3;
        u32x2 pk; pk.x = h0 | (h1 << 16); pk.y = h2 | (h3 << 16);
        *(u32x2*)(xb + (size_t)(i0 + r) * ED + k0 + c) = pk;
    }
    __syncthreads();
    {   // transposed store (4 consecutive i per thread)
        int r = tid >> 3;             // k within tile: 0..31
        int c = (tid & 7) * 4;        // i within tile: 0,4,..,28
        unsigned int h0 = tile[c + 0][r], h1 = tile[c + 1][r];
        unsigned int h2 = tile[c + 2][r], h3 = tile[c + 3][r];
        u32x2 pk; pk.x = h0 | (h1 << 16); pk.y = h2 | (h3 << 16);
        *(u32x2*)(xtb + (size_t)(k0 + r) * NS + i0 + c) = pk;
    }
}

// ---------------------------------------------------------------------------
// scores = Xb @ W^T + bias, fused hinge epilogue -> Gt (bf16 [NC][NS]).
// 2 waves/block (128 blocks for occupancy); B slab double-buffered in LDS via
// async copies; per K-step: 1 A fragment (global), 8 B fragments (LDS),
// 8 back-to-back WMMAs.
// ---------------------------------------------------------------------------
__global__ __launch_bounds__(64) void k_scores_margin(
    const unsigned short* __restrict__ xb,    // [NS][ED] bf16
    const unsigned short* __restrict__ wb,    // [NC][ED] bf16 (class-major)
    const float* __restrict__ bias,           // [NC]
    const int* __restrict__ labels,           // [NS]
    unsigned short* __restrict__ gt)          // out: [NC][NS] bf16
{
    __shared__ __align__(16) unsigned short bsl[2][NC][32];   // 16 KB
    __shared__ float smem[2][16][NC];                         // 16 KB
    const int tid  = threadIdx.x;
    const int lane = tid & 31;
    const int wv   = tid >> 5;
    const int row0 = (blockIdx.x * 2 + wv) * 16;
    const int l15  = lane & 15;
    const int kb   = (lane >> 4) * 8;

    v8f acc[8];
    #pragma unroll
    for (int t = 0; t < 8; ++t) acc[t] = v8f{0,0,0,0,0,0,0,0};

    stage_slab<64>(wb, 0, bsl[0], tid);
    async_copy_wait();
    __syncthreads();

    const u32x4* arow = (const u32x4*)(xb + (size_t)(row0 + l15) * ED);
    int buf = 0;
    for (int k0 = 0; k0 < ED; k0 += 32, buf ^= 1) {
        if (k0 + 32 < ED) stage_slab<64>(wb, k0 + 32, bsl[buf ^ 1], tid);
        __builtin_prefetch(arow + ((k0 + 256 + kb) >> 3), 0, 3);
        FragAB a;
        a.q[0] = arow[(k0 + kb) >> 3];
        a.q[1] = arow[(k0 + 16 + kb) >> 3];
        FragAB b[8];
        #pragma unroll
        for (int t = 0; t < 8; ++t) {
            const u32x4* br = (const u32x4*)(&bsl[buf][t * 16 + l15][0]);
            b[t].q[0] = br[kb >> 3];
            b[t].q[1] = br[(kb + 16) >> 3];
        }
        #pragma unroll
        for (int t = 0; t < 8; ++t) acc[t] = wmma_bf16(a.bf, b[t].bf, acc[t]);
        async_copy_wait();
        __syncthreads();
    }

    // D layout: VGPR e, lanes 0-15 -> M=e, lanes 16-31 -> M=8+e; N = lane&15
    const int mhi = (lane >> 4) * 8;
    #pragma unroll
    for (int t = 0; t < 8; ++t) {
        float bsc = bias[t * 16 + l15];
        #pragma unroll
        for (int e = 0; e < 8; ++e)
            smem[wv][mhi + e][t * 16 + l15] = acc[t][e] + bsc;
    }
    __syncthreads();

    if (lane < 16) {
        const int   r    = lane;
        const int   i    = row0 + r;
        const int   lab  = labels[i];
        const float corr = smem[wv][r][lab];
        int cnt = 0;
        for (int j = 0; j < NC; ++j) {
            float mg = smem[wv][r][j] - corr + 1.0f;
            cnt += (j != lab && mg > 0.0f) ? 1 : 0;
        }
        const float s = 1.0f / ((float)NS * (float)NC);   // mean() scale
        for (int j = 0; j < NC; ++j) {
            float mg = smem[wv][r][j] - corr + 1.0f;
            float g  = (j == lab) ? (-s * (float)cnt)
                                  : ((mg > 0.0f) ? s : 0.0f);
            gt[(size_t)j * NS + i] = (unsigned short)bf16_1(g);
        }
    }
}

// ---------------------------------------------------------------------------
// gradW^T = G^T @ X, fused: SGD update of Wt, bf16 re-pack into Wb, and
// (bk==0 blocks) bias update from the Gt row sums already streaming through
// the A fragments.  One wave per block: 16 classes x 64 embed-dims.
// ---------------------------------------------------------------------------
__global__ __launch_bounds__(32) void k_grad_update(
    const unsigned short* __restrict__ gt,    // [NC][NS] bf16
    const unsigned short* __restrict__ xtb,   // [ED][NS] bf16
    float* __restrict__ wt,                   // [NC][ED] f32 (updated)
    unsigned short* __restrict__ wbb,         // [NC][ED] bf16 (re-packed)
    float* __restrict__ bias)                 // [NC] (updated by bk==0 blocks)
{
    const int lane = threadIdx.x & 31;
    const int bn   = blockIdx.x >> 5;                 // class tile 0..7
    const int bk   = blockIdx.x & 31;                 // k-group   0..31
    const int n0   = bn * 16;
    const int k0   = bk * 64;                         // 64 k-dims per wave
    const int l15  = lane & 15;
    const int kb   = (lane >> 4) * 8;

    v8f acc[4];
    #pragma unroll
    for (int t = 0; t < 4; ++t) acc[t] = v8f{0,0,0,0,0,0,0,0};
    float bsum = 0.0f;                                // Gt row partial sums

    const u32x4* arow = (const u32x4*)(gt + (size_t)(n0 + l15) * NS);
    for (int i0 = 0; i0 < NS; i0 += 32) {
        FragAB a;
        a.q[0] = arow[(i0 + kb) >> 3];
        a.q[1] = arow[(i0 + 16 + kb) >> 3];
        FragAB b[4];
        #pragma unroll
        for (int t = 0; t < 4; ++t) {
            const u32x4* br = (const u32x4*)(xtb + (size_t)(k0 + t * 16 + l15) * NS);
            b[t].q[0] = br[(i0 + kb) >> 3];
            b[t].q[1] = br[(i0 + 16 + kb) >> 3];
        }
        #pragma unroll
        for (int t = 0; t < 4; ++t) acc[t] = wmma_bf16(a.bf, b[t].bf, acc[t]);
        if (bk == 0) {                                // free bias-grad partials
            #pragma unroll
            for (int c = 0; c < 8; ++c)
                bsum += bf16_to_f(a.u[c] & 0xFFFFu) + bf16_to_f(a.u[c] >> 16);
        }
    }

    const int mhi = (lane >> 4) * 8;
    #pragma unroll
    for (int t = 0; t < 4; ++t) {
        #pragma unroll
        for (int e = 0; e < 8; ++e) {
            int n = n0 + mhi + e;
            int k = k0 + t * 16 + l15;
            float* wp = wt + (size_t)n * ED + k;
            float w  = *wp;
            w = w - LRATE * (acc[t][e] + CREG * w);   // hinge grad + L2 reg
            *wp = w;
            wbb[(size_t)n * ED + k] = (unsigned short)bf16_1(w);  // fused pack
        }
    }

    if (bk == 0) {
        // lanes l and l+16 hold the two K-halves of Gt row n0+l
        float tot = bsum + __shfl_xor(bsum, 16, 32);
        if (lane < 16) bias[n0 + lane] -= LRATE * tot;
    }
}

// ---------------------------------------------------------------------------
// out = Q @ W^T + bias; Q converted f32->bf16 in-register; B slab from LDS
// (double-buffered, async copies), same pipeline as k_scores_margin.
// ---------------------------------------------------------------------------
__global__ __launch_bounds__(128) void k_query_gemm(
    const float* __restrict__ q,              // [NQ][ED] f32
    const unsigned short* __restrict__ wb,    // [NC][ED] bf16
    const float* __restrict__ bias,           // [NC]
    float* __restrict__ out)                  // [NQ][NC] f32
{
    __shared__ __align__(16) unsigned short bsl[2][NC][32];   // 16 KB
    const int tid  = threadIdx.x;
    const int lane = tid & 31;
    const int wv   = tid >> 5;
    const int row0 = (blockIdx.x * 4 + wv) * 16;
    const int l15  = lane & 15;
    const int kb   = (lane >> 4) * 8;

    v8f acc[8];
    #pragma unroll
    for (int t = 0; t < 8; ++t) acc[t] = v8f{0,0,0,0,0,0,0,0};

    stage_slab<128>(wb, 0, bsl[0], tid);
    async_copy_wait();
    __syncthreads();

    const float* arow = q + (size_t)(row0 + l15) * ED;
    int buf = 0;
    for (int k0 = 0; k0 < ED; k0 += 32, buf ^= 1) {
        if (k0 + 32 < ED) stage_slab<128>(wb, k0 + 32, bsl[buf ^ 1], tid);
        __builtin_prefetch(arow + k0 + 256 + kb, 0, 3);
        f32x4 f0 = *(const f32x4*)(arow + k0 + kb);
        f32x4 f1 = *(const f32x4*)(arow + k0 + kb + 4);
        f32x4 f2 = *(const f32x4*)(arow + k0 + 16 + kb);
        f32x4 f3 = *(const f32x4*)(arow + k0 + 16 + kb + 4);
        FragAB a;
        a.u[0] = bf16_2(f0.x, f0.y);  a.u[1] = bf16_2(f0.z, f0.w);
        a.u[2] = bf16_2(f1.x, f1.y);  a.u[3] = bf16_2(f1.z, f1.w);
        a.u[4] = bf16_2(f2.x, f2.y);  a.u[5] = bf16_2(f2.z, f2.w);
        a.u[6] = bf16_2(f3.x, f3.y);  a.u[7] = bf16_2(f3.z, f3.w);
        FragAB b[8];
        #pragma unroll
        for (int t = 0; t < 8; ++t) {
            const u32x4* br = (const u32x4*)(&bsl[buf][t * 16 + l15][0]);
            b[t].q[0] = br[kb >> 3];
            b[t].q[1] = br[(kb + 16) >> 3];
        }
        #pragma unroll
        for (int t = 0; t < 8; ++t) acc[t] = wmma_bf16(a.bf, b[t].bf, acc[t]);
        async_copy_wait();
        __syncthreads();
    }

    const int mhi = (lane >> 4) * 8;
    #pragma unroll
    for (int t = 0; t < 8; ++t) {
        float bsc = bias[t * 16 + l15];
        #pragma unroll
        for (int e = 0; e < 8; ++e)
            out[(size_t)(row0 + mhi + e) * NC + t * 16 + l15] = acc[t][e] + bsc;
    }
}

// ---------------------------------------------------------------------------
// Workspace layout (bytes, all 256B-aligned):
//   Xb  bf16[NS][ED]      @ 0          (16 MB)
//   XTb bf16[ED][NS]      @ 16777216   (16 MB)
//   Gt  bf16[NC][NS]      @ 33554432   ( 1 MB)
//   Wt  f32 [NC][ED]      @ 34603008   ( 1 MB)
//   Wb  bf16[NC][ED]      @ 35651584   (0.5 MB)
//   Bias f32[NC]          @ 36175872
// ---------------------------------------------------------------------------
extern "C" void kernel_launch(void* const* d_in, const int* in_sizes, int n_in,
                              void* d_out, int out_size, void* d_ws, size_t ws_size,
                              hipStream_t stream) {
    (void)in_sizes; (void)n_in; (void)out_size; (void)ws_size;
    const float* sup    = (const float*)d_in[0];
    const int*   labels = (const int*)d_in[1];
    const float* qry    = (const float*)d_in[2];
    float*       out    = (float*)d_out;

    char* ws = (char*)d_ws;
    unsigned short* Xb  = (unsigned short*)(ws);
    unsigned short* XTb = (unsigned short*)(ws + 16777216);
    unsigned short* Gt  = (unsigned short*)(ws + 33554432);
    float*          Wt  = (float*)         (ws + 34603008);
    unsigned short* Wb  = (unsigned short*)(ws + 35651584);
    float*          Bias= (float*)         (ws + 36175872);

    k_zero_w<<<(NC * ED + 255) / 256, 256, 0, stream>>>(Wt, Wb, Bias);
    k_cvt_support<<<(NS / 32) * (ED / 32), 256, 0, stream>>>(sup, Xb, XTb);

    for (int it = 0; it < MAX_ITER; ++it) {
        k_scores_margin<<<NS / 32, 64, 0, stream>>>(Xb, Wb, Bias, labels, Gt);
        k_grad_update<<<256, 32, 0, stream>>>(Gt, XTb, Wt, Wb, Bias);
    }

    k_query_gemm<<<NQ / 64, 128, 0, stream>>>(qry, Wb, Bias, out);
}
